// DyadBlock_83846351552721
// MI455X (gfx1250) — matrix operand order repaired and checked by reading, not compiled, not run on hardware
//
#include <hip/hip_runtime.h>

typedef __attribute__((ext_vector_type(16))) __bf16      v16bf;
typedef __attribute__((ext_vector_type(8)))  float        v8f;
typedef __attribute__((ext_vector_type(4)))  unsigned int v4u;

#define DYN 64
#define DIN 64
#define DON 64
#define BB  8192
#define BT  32            // B-columns per workgroup
#define LSTRIDE 72        // ushort stride per staged column (144B: 16B aligned, bank-spread)
#define LBUF (BT * LSTRIDE)

__device__ __forceinline__ unsigned short bf_bits(__bf16 v) {
  union { __bf16 b; unsigned short u; } c; c.b = v; return c.u;
}

// Packed-weight flat offset (in bf16 elements).
// Layout: [mat][dy][term(hi/lo)][kt][mt][lane] * 16 elems -> 32B contiguous per lane.
__device__ __forceinline__ size_t woff(int mat, int dy, int term, int kt, int mt, int lane) {
  return ((((((size_t)mat * DYN + dy) * 2 + term) * 2 + kt) * 4 + mt) * 32 + lane) * 16;
}

// ---------------------------------------------------------------------------
// Prep: repack w_upper/w_lower (fp32 [dy][do][di]) into bf16 hi/lo pairs laid
// out exactly as the 16-bit A-matrix 16x32 VGPR layout (ISA 7.12.2), so the
// main kernel's A fetch is one contiguous 32B load per lane.
// ---------------------------------------------------------------------------
__global__ __launch_bounds__(256) void dyad_prep(const float* __restrict__ wu,
                                                 const float* __restrict__ wl,
                                                 __bf16* __restrict__ wp) {
  int t    = blockIdx.x * 256 + threadIdx.x;   // 32768 threads total
  int lane = t & 31;
  int mt   = (t >> 5) & 3;
  int kt   = (t >> 7) & 1;
  int dy   = (t >> 8) & 63;
  int mat  = (t >> 14) & 1;
  const float* w = mat ? wl : wu;
  int m = lane & 15, h = lane >> 4;
  int o = mt * 16 + m;
  v16bf hv, lv;
#pragma unroll
  for (int e = 0; e < 16; ++e) {
    int v = e >> 1, p = e & 1;
    // A-matrix 16x32 (16-bit) lane layout: VGPR v<4 -> K = 2v+p+8h ; v>=4 -> K = 16+2(v-4)+p+8h
    int kl = (v < 4) ? (2 * v + p + 8 * h) : (16 + 2 * (v - 4) + p + 8 * h);
    int di = kt * 32 + kl;
    float val = w[((size_t)dy * DON + o) * DIN + di];
    __bf16 hi = (__bf16)val;
    __bf16 lo = (__bf16)(val - (float)hi);
    hv[e] = hi; lv[e] = lo;
  }
  *(v16bf*)(wp + woff(mat, dy, 0, kt, mt, lane)) = hv;
  *(v16bf*)(wp + woff(mat, dy, 1, kt, mt, lane)) = lv;
}

// ---------------------------------------------------------------------------
// Main: per 32-column B-tile. Pass 1: T = sum_j x_j. Pass 2: stream dy with
// running prefix P, suffix = T - P; bf16x3 split WMMA accumulation in fp32.
// ---------------------------------------------------------------------------
__global__ __launch_bounds__(256) void dyad_main(const float* __restrict__ x,
                                                 const __bf16* __restrict__ wp,
                                                 const float* __restrict__ bias,
                                                 float* __restrict__ out) {
  __shared__ __attribute__((aligned(32))) unsigned short smem[4 * LBUF];
  unsigned short* shS_h = smem;
  unsigned short* shS_l = smem + LBUF;
  unsigned short* shP_h = smem + 2 * LBUF;
  unsigned short* shP_l = smem + 3 * LBUF;

  const int t   = threadIdx.x;
  const int b   = t & (BT - 1);     // owned column within tile
  const int g   = t >> 5;           // wave id == di-group owner (8 rows each)
  const int dib = g * 8;
  const size_t col = (size_t)blockIdx.x * BT + b;

  // ---- Pass 1: column totals T (registers, 8 di-rows per thread) ----
  float T[8];
#pragma unroll
  for (int j = 0; j < 8; ++j) T[j] = 0.f;
  for (int dy = 0; dy < DYN; ++dy) {
    const float* xr = x + ((size_t)(dy * DIN + dib)) * BB + col;
#pragma unroll
    for (int j = 0; j < 8; ++j) T[j] += xr[(size_t)j * BB];
  }

  // wave roles: 4 M-tiles x 2 N-tiles of the 64x32 per-dy output
  const int lane = t & 31;
  const int mt = g & 3;
  const int nt = g >> 2;
  const int n  = lane & 15;
  const int h  = lane >> 4;
  const size_t colg = (size_t)blockIdx.x * BT + nt * 16 + n;

  float P[8];
#pragma unroll
  for (int j = 0; j < 8; ++j) P[j] = 0.f;

  for (int dy = 0; dy < DYN; ++dy) {
    // ---- stage S = T - P and P into LDS as bf16 hi/lo, B-operand layout ----
#pragma unroll
    for (int j = 0; j < 4; ++j) {
      float s0 = T[2 * j]     - P[2 * j];
      float s1 = T[2 * j + 1] - P[2 * j + 1];
      float p0 = P[2 * j], p1 = P[2 * j + 1];
      __bf16 s0h = (__bf16)s0; __bf16 s0l = (__bf16)(s0 - (float)s0h);
      __bf16 s1h = (__bf16)s1; __bf16 s1l = (__bf16)(s1 - (float)s1h);
      __bf16 p0h = (__bf16)p0; __bf16 p0l = (__bf16)(p0 - (float)p0h);
      __bf16 p1h = (__bf16)p1; __bf16 p1l = (__bf16)(p1 - (float)p1h);
      int u = b * LSTRIDE + dib + 2 * j;   // ushort index, 4B-aligned bytes
      *(unsigned int*)&shS_h[u] = (unsigned)bf_bits(s0h) | ((unsigned)bf_bits(s1h) << 16);
      *(unsigned int*)&shS_l[u] = (unsigned)bf_bits(s0l) | ((unsigned)bf_bits(s1l) << 16);
      *(unsigned int*)&shP_h[u] = (unsigned)bf_bits(p0h) | ((unsigned)bf_bits(p1h) << 16);
      *(unsigned int*)&shP_l[u] = (unsigned)bf_bits(p0l) | ((unsigned)bf_bits(p1l) << 16);
    }
    __syncthreads();

    // ---- WMMA: C = Wu*(T-P) + Wl*P via bf16x3 split, fp32 accumulate ----
    v8f c = {};
#pragma unroll
    for (int kt = 0; kt < 2; ++kt) {
      const v16bf wu_h = *(const v16bf*)(wp + woff(0, dy, 0, kt, mt, lane));
      const v16bf wu_l = *(const v16bf*)(wp + woff(0, dy, 1, kt, mt, lane));
      const v16bf wl_h = *(const v16bf*)(wp + woff(1, dy, 0, kt, mt, lane));
      const v16bf wl_l = *(const v16bf*)(wp + woff(1, dy, 1, kt, mt, lane));
      const int bo = (nt * 16 + n) * LSTRIDE + kt * 32 + h * 8;  // 16B-aligned bytes
      union { v4u u[2]; v16bf v; } Sh, Sl, Ph, Pl;
      Sh.u[0] = *(const v4u*)&shS_h[bo]; Sh.u[1] = *(const v4u*)&shS_h[bo + 16];
      Sl.u[0] = *(const v4u*)&shS_l[bo]; Sl.u[1] = *(const v4u*)&shS_l[bo + 16];
      Ph.u[0] = *(const v4u*)&shP_h[bo]; Ph.u[1] = *(const v4u*)&shP_h[bo + 16];
      Pl.u[0] = *(const v4u*)&shP_l[bo]; Pl.u[1] = *(const v4u*)&shP_l[bo + 16];
      c = __builtin_amdgcn_wmma_f32_16x16x32_bf16(false, wu_h, false, Sh.v, (short)0, c, false, false);
      c = __builtin_amdgcn_wmma_f32_16x16x32_bf16(false, wu_h, false, Sl.v, (short)0, c, false, false);
      c = __builtin_amdgcn_wmma_f32_16x16x32_bf16(false, wu_l, false, Sh.v, (short)0, c, false, false);
      c = __builtin_amdgcn_wmma_f32_16x16x32_bf16(false, wl_h, false, Ph.v, (short)0, c, false, false);
      c = __builtin_amdgcn_wmma_f32_16x16x32_bf16(false, wl_h, false, Pl.v, (short)0, c, false, false);
      c = __builtin_amdgcn_wmma_f32_16x16x32_bf16(false, wl_l, false, Ph.v, (short)0, c, false, false);
    }

    // ---- bias add + store (C/D layout: VGPR r -> row r + 8h, col n) ----
#pragma unroll
    for (int r = 0; r < 8; ++r) {
      int row = dy * DON + mt * 16 + 8 * h + r;
      out[(size_t)row * BB + colg] = c[r] + bias[row];
    }

    // ---- advance running prefix (x re-read is L2-resident) ----
    const float* xr = x + ((size_t)(dy * DIN + dib)) * BB + col;
#pragma unroll
    for (int j = 0; j < 8; ++j) P[j] += xr[(size_t)j * BB];

    __syncthreads();
  }
}

extern "C" void kernel_launch(void* const* d_in, const int* in_sizes, int n_in,
                              void* d_out, int out_size, void* d_ws, size_t ws_size,
                              hipStream_t stream) {
  (void)in_sizes; (void)n_in; (void)out_size; (void)ws_size;
  const float* x    = (const float*)d_in[0];
  const float* wu   = (const float*)d_in[1];
  const float* wl   = (const float*)d_in[2];
  const float* bias = (const float*)d_in[3];
  float* out = (float*)d_out;
  __bf16* wp = (__bf16*)d_ws;   // 2 MiB packed weights

  dyad_prep<<<128, 256, 0, stream>>>(wu, wl, wp);
  dyad_main<<<BB / BT, 256, 0, stream>>>(x, wp, bias, out);
}
